// DeltaNet_31877247271483
// MI455X (gfx1250) — matrix operand order repaired and checked by reading, not compiled, not run on hardware
//
#include <hip/hip_runtime.h>

// ---------------- problem constants (B,H,L,D,chunk from reference) ----------
#define BB   4
#define HH   8
#define LL   4096
#define DK   64
#define DV   64
#define CC   32
#define NCH  (LL / CC)     // 128 chunks
#define BH   (BB * HH)     // 32 (b,h) streams

// LDS strides (in u16 elements)
#define S64  80            // padded 64-col tiles written element-wise (Sb)
#define S32  40            // padded 32-col tiles (kT, upT, Pb, T)
#define SQT  64            // unpadded 64-col tiles landed contiguously by TDM

typedef __attribute__((ext_vector_type(16))) __bf16 v16bf;
typedef __attribute__((ext_vector_type(8)))  float  v8f;
typedef unsigned int u32x4 __attribute__((ext_vector_type(4)));
typedef int          i32x8 __attribute__((ext_vector_type(8)));
typedef int          i32x4 __attribute__((ext_vector_type(4)));

__device__ __forceinline__ unsigned short f32_to_bf16(float f) {
  unsigned int u = __float_as_uint(f);
  u += 0x7FFFu + ((u >> 16) & 1u);       // round-to-nearest-even
  return (unsigned short)(u >> 16);
}

// A-operand fragment (16xK tile slice, row-major LDS, base -> [row0][k0]).
// ISA 16-bit A layout: lane half h, element e: k = (e/8)*16 + h*8 + e%8
// -> two contiguous 8-element (16B) runs per lane.
__device__ __forceinline__ v16bf frag_a(const unsigned short* base, int stride) {
  const int lane = threadIdx.x & 31;
  const unsigned short* p = base + (lane & 15) * stride + (lane >> 4) * 8;
  union { v16bf v; unsigned short u[16]; } f;
#pragma unroll
  for (int e = 0; e < 8; ++e) f.u[e] = p[e];
#pragma unroll
  for (int e = 0; e < 8; ++e) f.u[8 + e] = p[16 + e];
  return f.v;
}

// B-operand fragment from an n-major ("Bt") tile: Bt[n][k], base -> [n0][k0].
// ISA 16-bit B layout: lane half h, element e: k = (e/8)*16 + h*8 + e%8 (matches A).
__device__ __forceinline__ v16bf frag_bt(const unsigned short* base, int stride) {
  const int lane = threadIdx.x & 31;
  const unsigned short* p = base + (lane & 15) * stride + (lane >> 4) * 8;
  union { v16bf v; unsigned short u[16]; } f;
#pragma unroll
  for (int e = 0; e < 8; ++e) f.u[e] = p[e];
#pragma unroll
  for (int e = 0; e < 8; ++e) f.u[8 + e] = p[16 + e];
  return f.v;
}

__device__ __forceinline__ v8f wmma_bf16(v16bf a, v16bf b, v8f c) {
  // D(16x16,f32) = A(16x32,bf16) * B(32x16,bf16) + C
  return __builtin_amdgcn_wmma_f32_16x16x32_bf16(false, a, false, b, (short)0, c,
                                                 false, false);
}

// ---------------------------------------------------------------------------
// Tensor Data Mover: 1-D contiguous global -> LDS DMA (8-byte elements).
// D# bitfields per CDNA5 ISA:
//   g0: [1:0]=count=1 | [63:32]=lds_addr | [120:64]=global_addr | [127:126]=2
//   g1: [17:16]=data_size(3=8B) | [79:48]=tensor_dim0 | [127:112]=tile_dim0
// ---------------------------------------------------------------------------
#if __has_builtin(__builtin_amdgcn_tensor_load_to_lds)
#define HAVE_TDM 1
__device__ __forceinline__ void tdm_load_1d(void* lds_ptr, const void* gptr,
                                            unsigned int elems8B) {
  // flat LDS address truncates to the LDS byte offset (ISA §10.2 LDS aperture)
  unsigned int lds_off = (unsigned int)(unsigned long long)lds_ptr;
  unsigned long long ga = (unsigned long long)gptr;
  u32x4 g0;
  g0[0] = 1u;                                                     // count=1
  g0[1] = lds_off;                                                // lds_addr
  g0[2] = (unsigned int)ga;                                       // addr[31:0]
  g0[3] = (unsigned int)((ga >> 32) & 0x01FFFFFFull) | (2u << 30);// addr[56:32]|type=2
  i32x8 g1;
  g1[0] = (int)(3u << 16);                      // wg_mask=0, data_size=8B
  g1[1] = (int)((elems8B & 0xFFFFu) << 16);     // tensor_dim0[15:0]
  g1[2] = (int)((elems8B >> 16) & 0xFFFFu);     // tensor_dim0[31:16]
  g1[3] = (int)((elems8B & 0xFFFFu) << 16);     // tile_dim0
  g1[4] = 0; g1[5] = 0; g1[6] = 0; g1[7] = 0;   // higher dims unused
  i32x4 z = {0, 0, 0, 0};
#if __clang_major__ >= 23
  i32x8 z8 = {0, 0, 0, 0, 0, 0, 0, 0};          // 6-arg toolchain form
  __builtin_amdgcn_tensor_load_to_lds(g0, g1, z, z, z8, 0);
#else
  __builtin_amdgcn_tensor_load_to_lds(g0, g1, z, z, 0);           // ROCm 7.2 form
#endif
}
#else
#define HAVE_TDM 0
#endif

// ============================================================================
// Kernel A: per-chunk preprocessing (fully parallel over BH*NCH chunks)
//   qn = l2norm(q); kn = l2norm(k); kb = kn*beta; vb = v*beta
//   T  = forward-substitution inverse (f32) of the WY triangular system
//   u = T @ vb (f32 out), w = T @ kb (bf16 out)
// ============================================================================
__global__ __launch_bounds__(256) void deltanet_chunk_kernel(
    const float* __restrict__ g_q, const float* __restrict__ g_k,
    const float* __restrict__ g_v, const float* __restrict__ g_beta,
    unsigned short* __restrict__ ws_qn, unsigned short* __restrict__ ws_kn,
    unsigned short* __restrict__ ws_w, float* __restrict__ ws_u)
{
  __shared__ unsigned short kn_s [CC * S64];   // kn row-major  (Bt for kb.knT)
  __shared__ unsigned short kb_s [CC * S64];   // kb row-major  (A operand)
  __shared__ unsigned short kbT_s[DK * S32];   // kb transposed (Bt for T@kb)
  __shared__ unsigned short vbT_s[DV * S32];   // vb transposed (Bt for T@vb)
  __shared__ unsigned short T_s  [CC * S32];   // T bf16        (A operand)
  __shared__ float attn_s[CC * 33];
  __shared__ float red_s [CC * 8];
  __shared__ float scl_s [CC];

  const int tid  = threadIdx.x;
  const int wv   = tid >> 5;
  const int lane = tid & 31;
  const int bh   = blockIdx.x / NCH;
  const int t    = blockIdx.x % NCH;
  const size_t goff = ((size_t)bh * LL + (size_t)t * CC) * DK;   // into q/k/v
  const size_t coff = (size_t)blockIdx.x * (size_t)(CC * DK);    // into ws

  const int r  = tid >> 3;            // chunk row 0..31 (8 threads per row)
  const int c0 = (tid & 7) * 8;       // 8 contiguous columns per thread
  const float br = g_beta[(size_t)bh * LL + (size_t)t * CC + r];

  // ---- q: l2norm, emit bf16 qn ----
  float qv[8];
  {
    const float* p = g_q + goff + r * DK + c0;
    float ss = 0.f;
#pragma unroll
    for (int e = 0; e < 8; ++e) { qv[e] = p[e]; ss += qv[e] * qv[e]; }
    red_s[r * 8 + (tid & 7)] = ss;
  }
  __syncthreads();
  if (tid < CC) {
    float s = 0.f;
#pragma unroll
    for (int e = 0; e < 8; ++e) s += red_s[tid * 8 + e];
    scl_s[tid] = rsqrtf(s + 1e-6f);
  }
  __syncthreads();
  {
    const float sc = scl_s[r];
    unsigned short* o = ws_qn + coff + r * DK + c0;
#pragma unroll
    for (int e = 0; e < 8; ++e) o[e] = f32_to_bf16(qv[e] * sc);
  }
  __syncthreads();

  // ---- k: l2norm, emit kn (bf16), stage kn/kb/kbT in LDS ----
  float kv[8];
  {
    const float* p = g_k + goff + r * DK + c0;
    float ss = 0.f;
#pragma unroll
    for (int e = 0; e < 8; ++e) { kv[e] = p[e]; ss += kv[e] * kv[e]; }
    red_s[r * 8 + (tid & 7)] = ss;
  }
  __syncthreads();
  if (tid < CC) {
    float s = 0.f;
#pragma unroll
    for (int e = 0; e < 8; ++e) s += red_s[tid * 8 + e];
    scl_s[tid] = rsqrtf(s + 1e-6f);
  }
  __syncthreads();
  {
    const float sc = scl_s[r];
    unsigned short* o = ws_kn + coff + r * DK + c0;
#pragma unroll
    for (int e = 0; e < 8; ++e) {
      float kn  = kv[e] * sc;
      unsigned short knb = f32_to_bf16(kn);
      unsigned short kbb = f32_to_bf16(kn * br);
      o[e] = knb;
      kn_s [r * S64 + c0 + e]   = knb;
      kb_s [r * S64 + c0 + e]   = kbb;
      kbT_s[(c0 + e) * S32 + r] = kbb;
    }
  }
  // ---- v: vb = v*beta, stage transposed ----
  {
    const float* p = g_v + goff + r * DK + c0;
#pragma unroll
    for (int e = 0; e < 8; ++e) vbT_s[(c0 + e) * S32 + r] = f32_to_bf16(p[e] * br);
  }
  __syncthreads();

  // ---- A = -tril(kb @ knT, -1)  (32x32, K=64) : waves 0..3, one tile each ----
  if (wv < 4) {
    const int mi = wv >> 1, ni = wv & 1;
    v8f acc = {0.f, 0.f, 0.f, 0.f, 0.f, 0.f, 0.f, 0.f};
    acc = wmma_bf16(frag_a(kb_s + mi * 16 * S64,      S64),
                    frag_bt(kn_s + ni * 16 * S64,      S64), acc);
    acc = wmma_bf16(frag_a(kb_s + mi * 16 * S64 + 32, S64),
                    frag_bt(kn_s + ni * 16 * S64 + 32, S64), acc);
    const int n = lane & 15, mh = (lane >> 4) * 8;
#pragma unroll
    for (int vv = 0; vv < 8; ++vv) {
      int row = mi * 16 + mh + vv, col = ni * 16 + n;
      attn_s[row * 33 + col] = (col < row) ? -acc[vv] : 0.0f;
    }
  }
  __syncthreads();

  // ---- forward substitution (f32, 31 serialized row updates) ----
  for (int i = 1; i < CC; ++i) {
    float s = 0.f;
    if (tid < CC) {
      for (int j = 0; j < i; ++j) s += attn_s[i * 33 + j] * attn_s[j * 33 + tid];
    }
    __syncthreads();
    if (tid < i) attn_s[i * 33 + tid] += s;
    __syncthreads();
  }
  if (tid < CC) attn_s[tid * 33 + tid] = 1.0f;   // + I
  __syncthreads();
  {
    const int rr = tid >> 3, cc = (tid & 7) * 4;   // 1024 elems / 256 thr
#pragma unroll
    for (int e = 0; e < 4; ++e)
      T_s[rr * S32 + cc + e] = f32_to_bf16(attn_s[rr * 33 + cc + e]);
  }
  __syncthreads();

  // ---- u = T @ vb (f32 out), w = T @ kb (bf16 out); 8 tiles over 8 waves ----
  {
    const int mi = wv >> 2, ni = wv & 3;
    v16bf ta = frag_a(T_s + mi * 16 * S32, S32);     // shared A fragment (K=32)
    v8f au = {0.f, 0.f, 0.f, 0.f, 0.f, 0.f, 0.f, 0.f};
    v8f aw = {0.f, 0.f, 0.f, 0.f, 0.f, 0.f, 0.f, 0.f};
    au = wmma_bf16(ta, frag_bt(vbT_s + ni * 16 * S32, S32), au);
    aw = wmma_bf16(ta, frag_bt(kbT_s + ni * 16 * S32, S32), aw);
    const int n = lane & 15, mh = (lane >> 4) * 8;
#pragma unroll
    for (int vv = 0; vv < 8; ++vv) {
      int row = mi * 16 + mh + vv, col = ni * 16 + n;
      ws_u[coff + row * DK + col] = au[vv];
      ws_w[coff + row * DK + col] = f32_to_bf16(aw[vv]);
    }
  }
}

// ============================================================================
// Kernel B: sequential chunk scan per (b,h); S lives in f32 WMMA accumulators.
// Per-chunk operand tiles are DMA'd into LDS by the Tensor Data Mover.
//   u' = u - w@S ; P = tril(q@kT) ; o = q@S + P@u' ; S += kT@u'
// ============================================================================
__global__ __launch_bounds__(256) void deltanet_scan_kernel(
    const unsigned short* __restrict__ ws_qn, const unsigned short* __restrict__ ws_kn,
    const unsigned short* __restrict__ ws_w,  const float* __restrict__ ws_u,
    float* __restrict__ g_out)
{
  __shared__ unsigned short qA [CC * SQT];   // q  row-major (A; also for P) [TDM]
  __shared__ unsigned short kB [CC * SQT];   // k  row-major (Bt for q@kT)   [TDM]
  __shared__ unsigned short wA [CC * SQT];   // w  row-major (A)             [TDM]
  __shared__ float          uF [CC * DK];    // u  f32                       [TDM]
  __shared__ unsigned short kT [DK * S32];   // k  transposed (A for S update)
  __shared__ unsigned short upT[DV * S32];   // u' transposed (Bt)
  __shared__ unsigned short Pb [CC * S32];   // P bf16 (A)
  __shared__ unsigned short Sb [DV * S64];   // S^T bf16 [dv][dk] (Bt for *@S)

  const int tid = threadIdx.x, wv = tid >> 5, lane = tid & 31;
  const int n = lane & 15, mh = (lane >> 4) * 8;
  const int bh = blockIdx.x;
  const int mi = wv >> 2, ni = wv & 3;     // 2x4 tiling of 32x64 outputs
  const int sm = wv & 3;                   // S: dk tile (4)
  const int sn0 = (wv >> 2) * 2;           // S: dv tile pair (2 per wave)

  v8f Sacc0 = {0.f, 0.f, 0.f, 0.f, 0.f, 0.f, 0.f, 0.f};
  v8f Sacc1 = {0.f, 0.f, 0.f, 0.f, 0.f, 0.f, 0.f, 0.f};
  for (int i = tid; i < DV * S64; i += 256) Sb[i] = 0;
  __syncthreads();

  float* g_o = g_out + (size_t)bh * LL * DV;
  float* g_S = g_out + (size_t)BB * HH * LL * DV + (size_t)bh * DK * DV;

  const int r = tid >> 3, c0 = (tid & 7) * 8;

  for (int t = 0; t < NCH; ++t) {
    const size_t coff = ((size_t)bh * NCH + t) * (size_t)(CC * DK);

    // -------- stage chunk operands into LDS --------
#if HAVE_TDM
    if (wv == 0) {
      tdm_load_1d(qA, ws_qn + coff, (CC * DK * 2) / 8);   // 4KB bf16
      tdm_load_1d(kB, ws_kn + coff, (CC * DK * 2) / 8);   // 4KB bf16
      tdm_load_1d(wA, ws_w  + coff, (CC * DK * 2) / 8);   // 4KB bf16
      tdm_load_1d(uF, ws_u  + coff, (CC * DK * 4) / 8);   // 8KB f32
      __builtin_amdgcn_s_wait_tensorcnt(0);
    }
    if (t + 1 < NCH) {                  // warm L2 for the next chunk
      const size_t noff = coff + (size_t)(CC * DK);
      __builtin_prefetch(ws_qn + noff + tid * 8, 0, 1);
      __builtin_prefetch(ws_kn + noff + tid * 8, 0, 1);
      __builtin_prefetch(ws_w  + noff + tid * 8, 0, 1);
      __builtin_prefetch(ws_u  + noff + tid * 8, 0, 1);
    }
    __syncthreads();                    // TDM-landed tiles visible to all waves
    {                                   // build kT (transposed) from kB tile
      union { uint4 v; unsigned short s[8]; } d;
      d.v = *reinterpret_cast<const uint4*>(&kB[r * SQT + c0]);
#pragma unroll
      for (int e = 0; e < 8; ++e) kT[(c0 + e) * S32 + r] = d.s[e];
    }
#else
    {
      union { uint4 v; unsigned short s[8]; } d;
      d.v = *reinterpret_cast<const uint4*>(ws_qn + coff + r * DK + c0);
      *reinterpret_cast<uint4*>(&qA[r * SQT + c0]) = d.v;
      d.v = *reinterpret_cast<const uint4*>(ws_kn + coff + r * DK + c0);
      *reinterpret_cast<uint4*>(&kB[r * SQT + c0]) = d.v;
#pragma unroll
      for (int e = 0; e < 8; ++e) kT[(c0 + e) * S32 + r] = d.s[e];
      d.v = *reinterpret_cast<const uint4*>(ws_w + coff + r * DK + c0);
      *reinterpret_cast<uint4*>(&wA[r * SQT + c0]) = d.v;
      const float4* pu = reinterpret_cast<const float4*>(ws_u + coff + r * DK + c0);
      *reinterpret_cast<float4*>(&uF[r * DK + c0])     = pu[0];
      *reinterpret_cast<float4*>(&uF[r * DK + c0 + 4]) = pu[1];
    }
#endif
    __syncthreads();

    // -------- u' = u - w @ S  -> upT (bf16, [dv][c]) : 8 tiles / 8 waves ----
    {
      v8f acc = {0.f, 0.f, 0.f, 0.f, 0.f, 0.f, 0.f, 0.f};
      acc = wmma_bf16(frag_a(wA + mi * 16 * SQT,      SQT),
                      frag_bt(Sb + ni * 16 * S64,      S64), acc);
      acc = wmma_bf16(frag_a(wA + mi * 16 * SQT + 32, SQT),
                      frag_bt(Sb + ni * 16 * S64 + 32, S64), acc);
#pragma unroll
      for (int vv = 0; vv < 8; ++vv) {
        int row = mi * 16 + mh + vv, col = ni * 16 + n;
        upT[col * S32 + row] = f32_to_bf16(uF[row * DK + col] - acc[vv]);
      }
    }
    // -------- P = tril(q @ kT) : 4 tiles, waves 0..3 --------
    if (wv < 4) {
      const int pm = wv >> 1, pn = wv & 1;
      v8f acc = {0.f, 0.f, 0.f, 0.f, 0.f, 0.f, 0.f, 0.f};
      acc = wmma_bf16(frag_a(qA + pm * 16 * SQT,      SQT),
                      frag_bt(kB + pn * 16 * SQT,      SQT), acc);
      acc = wmma_bf16(frag_a(qA + pm * 16 * SQT + 32, SQT),
                      frag_bt(kB + pn * 16 * SQT + 32, SQT), acc);
#pragma unroll
      for (int vv = 0; vv < 8; ++vv) {
        int row = pm * 16 + mh + vv, col = pn * 16 + n;
        Pb[row * S32 + col] = f32_to_bf16((col <= row) ? acc[vv] : 0.0f);
      }
    }
    __syncthreads();

    // -------- o = q @ S + P @ u' : 8 tiles / 8 waves --------
    {
      v8f acc = {0.f, 0.f, 0.f, 0.f, 0.f, 0.f, 0.f, 0.f};
      acc = wmma_bf16(frag_a(qA + mi * 16 * SQT,      SQT),
                      frag_bt(Sb + ni * 16 * S64,      S64), acc);
      acc = wmma_bf16(frag_a(qA + mi * 16 * SQT + 32, SQT),
                      frag_bt(Sb + ni * 16 * S64 + 32, S64), acc);
      acc = wmma_bf16(frag_a(Pb + mi * 16 * S32,       S32),
                      frag_bt(upT + ni * 16 * S32,     S32), acc);
      float* po = g_o + (size_t)t * CC * DV;
#pragma unroll
      for (int vv = 0; vv < 8; ++vv)
        po[(mi * 16 + mh + vv) * DV + ni * 16 + n] = acc[vv];
    }
    __syncthreads();   // everyone done reading Sb / qA / upT for this chunk

    // -------- S += kT @ u' : 16 tiles, 2 per wave (persistent f32 accum) ----
    {
      v16bf ka = frag_a(kT + sm * 16 * S32, S32);
      Sacc0 = wmma_bf16(ka, frag_bt(upT + sn0 * 16 * S32,       S32), Sacc0);
      Sacc1 = wmma_bf16(ka, frag_bt(upT + (sn0 + 1) * 16 * S32, S32), Sacc1);
#pragma unroll
      for (int vv = 0; vv < 8; ++vv) {
        int kk = sm * 16 + mh + vv;
        Sb[(sn0 * 16 + n) * S64 + kk]       = f32_to_bf16(Sacc0[vv]);
        Sb[((sn0 + 1) * 16 + n) * S64 + kk] = f32_to_bf16(Sacc1[vv]);
      }
    }
    __syncthreads();
  }

  // final state S (B,H,dk,dv) f32, concatenated after o in d_out
#pragma unroll
  for (int vv = 0; vv < 8; ++vv) {
    int kk = sm * 16 + mh + vv;
    g_S[kk * DV + sn0 * 16 + n]       = Sacc0[vv];
    g_S[kk * DV + (sn0 + 1) * 16 + n] = Sacc1[vv];
  }
}

// ============================================================================
extern "C" void kernel_launch(void* const* d_in, const int* in_sizes, int n_in,
                              void* d_out, int out_size, void* d_ws, size_t ws_size,
                              hipStream_t stream)
{
  (void)in_sizes; (void)n_in; (void)out_size; (void)ws_size;
  const float* q    = (const float*)d_in[0];
  const float* k    = (const float*)d_in[1];
  const float* v    = (const float*)d_in[2];
  const float* beta = (const float*)d_in[3];
  // d_in[4] = chunk_size (int, ==32; layout specialized at compile time)

  const size_t nchunks     = (size_t)BH * NCH;   // 4096
  const size_t chunk_elems = (size_t)CC * DK;    // 2048
  unsigned short* ws_qn = (unsigned short*)d_ws;
  unsigned short* ws_kn = ws_qn + nchunks * chunk_elems;
  unsigned short* ws_w  = ws_kn + nchunks * chunk_elems;
  float*          ws_u  = (float*)(ws_w + nchunks * chunk_elems);

  deltanet_chunk_kernel<<<dim3(BH * NCH), dim3(256), 0, stream>>>(
      q, k, v, beta, ws_qn, ws_kn, ws_w, ws_u);
  deltanet_scan_kernel<<<dim3(BH), dim3(256), 0, stream>>>(
      ws_qn, ws_kn, ws_w, ws_u, (float*)d_out);
}
